// FractalSSM_59236188946847
// MI455X (gfx1250) — matrix operand order repaired and checked by reading, not compiled, not run on hardware
//
#include <hip/hip_runtime.h>
#include <hip/hip_bf16.h>
#include <math.h>

// ---------------------------------------------------------------------------
// FractalSSM forward for MI455X (gfx1250, wave32, WMMA + async-to-LDS).
//   All GEMM operands are zero-padded in workspace to tile multiples so the
//   staging loop has ONLY full-tile b128 vector paths (no scalar fallbacks).
//   A-tile staging uses GLOBAL_LOAD_ASYNC_TO_LDS_B128 (ASYNCcnt) on gfx1250.
// ---------------------------------------------------------------------------

typedef __attribute__((ext_vector_type(16))) __bf16 v16bf;
typedef __attribute__((ext_vector_type(8)))  __bf16 v8bf;
typedef __attribute__((ext_vector_type(8)))  float  v8f;

#define D_   768
#define L_   1024
#define B_   4
#define N_   16
#define R_   48
#define XPW  112   /* R + 4N (logical) */
#define XPWP 128   /* padded to tile multiple */
#define KDTP 64    /* dt GEMM K, padded from 48 */
#define MTOT 4096  /* B*L */

// ---- gfx1250 async global->LDS staging (ASYNCcnt) ----
#if defined(__gfx1250__)
#define ASYNC_LDS_LOAD_B128(ldst, gsrc)                                        \
    __asm__ volatile("global_load_async_to_lds_b128 %0, %1, off"               \
                     :: "v"((unsigned)(uintptr_t)(ldst)),                      \
                        "v"((unsigned long long)(uintptr_t)(gsrc))             \
                     : "memory")
#define ASYNC_WAIT() __asm__ volatile("s_wait_asynccnt 0x0" ::: "memory")
#define HAVE_ASYNC 1
#endif

__global__ __launch_bounds__(256)
void pad_convert_kernel(const float* __restrict__ src, __bf16* __restrict__ dst,
                        int srcRows, int srcCols, int dstCols, int total)
{
    int i = blockIdx.x * 256 + threadIdx.x;
    if (i >= total) return;
    int r = i / dstCols;
    int c = i - r * dstCols;
    dst[i] = (r < srcRows && c < srcCols) ? (__bf16)src[r * srcCols + c]
                                          : (__bf16)0.0f;
}

__device__ __forceinline__ float silu_f(float v) {
    return v / (1.0f + __expf(-v));
}

// EPI 0: in_proj  (N=1536): silu; cols<768 -> x_in f32+bf16, cols>=768 -> zsilu
// EPI 1: x_proj   (N=128 padded): xp f32 (col<112) + xp bf16 (all cols)
// EPI 2: dt_proj  (N=768): +bias, softplus, dt f32
// EPI 3: out_proj (N=768): +bias, f32 -> d_out
template<int EPI>
__global__ __launch_bounds__(256)
void gemm_bf16_wmma(const __bf16* __restrict__ A, int lda,
                    const __bf16* __restrict__ W, int ldw,
                    int K,
                    const float*  __restrict__ bias,
                    float*        __restrict__ out0,
                    __bf16*       __restrict__ out1,
                    float*        __restrict__ out2)
{
    __shared__ __bf16 As[128 * 32];   // [row][k]             8 KB
    __shared__ __bf16 Bs[64 * 32];    // [n][k] (transposed)  4 KB

    const int t     = threadIdx.x;
    const int wave  = t >> 5;
    const int lane  = t & 31;
    const int ln    = lane & 15;
    const int lh    = lane >> 4;            // 0/1: K-half select
    const int rowBase = blockIdx.y * 128;
    const int nBase   = blockIdx.x * 64;
    const int waveRow = rowBase + wave * 16;

    const v8f vzero = {0.f, 0.f, 0.f, 0.f, 0.f, 0.f, 0.f, 0.f};
    v8f acc[4];
    #pragma unroll
    for (int i = 0; i < 4; ++i) acc[i] = vzero;

    // all tiles are full: K % 32 == 0, N % 64 == 0, M % 128 == 0
    const int arow = t >> 1;              // 0..127
    const int ako  = (t & 1) * 16;        // 0 / 16
    const int krow = t >> 3;              // 0..31
    const int no   = (t & 7) * 8;         // 0..56

    for (int k0 = 0; k0 < K; k0 += 32) {
        // ---- stage A tile 128x32: async global->LDS, 2x16B per thread ----
        const __bf16* asrc = A + (size_t)(rowBase + arow) * lda + k0 + ako;
        __bf16*       adst = &As[arow * 32 + ako];
#ifdef HAVE_ASYNC
        ASYNC_LDS_LOAD_B128(adst,     asrc);
        ASYNC_LDS_LOAD_B128(adst + 8, asrc + 8);
#else
        *(v8bf*)adst       = *(const v8bf*)asrc;
        *(v8bf*)(adst + 8) = *(const v8bf*)(asrc + 8);
#endif
        // ---- stage W tile 32x64 -> LDS transposed [n][k] ----
        {
            v8bf wv = *(const v8bf*)(W + (size_t)(k0 + krow) * ldw + nBase + no);
            #pragma unroll
            for (int j = 0; j < 8; ++j) Bs[(no + j) * 32 + krow] = wv[j];
        }
        // prefetch next K tile (global_prefetch_b8)
        if (k0 + 32 < K) {
            __builtin_prefetch(A + (size_t)(rowBase + arow) * lda + k0 + 32, 0, 1);
            __builtin_prefetch(W + (size_t)(k0 + 32 + krow) * ldw + nBase, 0, 1);
        }
#ifdef HAVE_ASYNC
        ASYNC_WAIT();
#endif
        __syncthreads();

        // ---- A fragment: lane ln = row m, K = {lh*8..+7, lh*8+16..+7} ----
        const int m = wave * 16 + ln;
        v8bf alo = *(const v8bf*)&As[m * 32 + lh * 8];
        v8bf ahi = *(const v8bf*)&As[m * 32 + lh * 8 + 16];
        v16bf afrag = __builtin_shufflevector(alo, ahi,
            0,1,2,3,4,5,6,7,8,9,10,11,12,13,14,15);

        #pragma unroll
        for (int nt = 0; nt < 4; ++nt) {
            const int col = nt * 16 + ln;
            const int ks  = lh * 8;
            v8bf blo = *(const v8bf*)&Bs[col * 32 + ks];
            v8bf bhi = *(const v8bf*)&Bs[col * 32 + ks + 16];
            v16bf bfrag = __builtin_shufflevector(blo, bhi,
                0,1,2,3,4,5,6,7,8,9,10,11,12,13,14,15);
            acc[nt] = __builtin_amdgcn_wmma_f32_16x16x32_bf16(
                false, afrag, false, bfrag, (short)0, acc[nt], false, false);
        }
        __syncthreads();
    }

    // ---- epilogue: lane holds col = ln, rows r + lh*8 ----
    #pragma unroll
    for (int nt = 0; nt < 4; ++nt) {
        const int col = nBase + nt * 16 + ln;
        #pragma unroll
        for (int r = 0; r < 8; ++r) {
            const int row = waveRow + lh * 8 + r;
            float v = acc[nt][r];
            if (EPI == 0) {
                float s = silu_f(v);
                if (col < D_) {
                    out0[(size_t)row * D_ + col] = s;
                    out1[(size_t)row * D_ + col] = (__bf16)s;
                } else {
                    out2[(size_t)row * D_ + (col - D_)] = s;
                }
            } else if (EPI == 1) {
                // padded W cols [112,128) are zero => v==0 there; bf16 store is safe
                if (col < XPW) out0[(size_t)row * XPW + col] = v;
                out1[(size_t)row * XPWP + col] = (__bf16)v;
            } else if (EPI == 2) {
                float u = v + bias[col];
                float sp = (u > 20.0f) ? u : log1pf(__expf(u));
                out0[(size_t)row * D_ + col] = sp;
            } else {
                out0[(size_t)row * D_ + col] = v + bias[col];
            }
        }
    }
}

// ---------------------------------------------------------------------------
// Scan: thread = one (b,d,n). lanes [0..15] of each half-wave share one (b,d).
// Polar math strength-reduced:  cm*cos(atan2(h1,h0)) == h0 * (cm/mag),
// tanh(x) = 1 - 2/(e^{2x}+1)  -> one v_exp_f32 per squash.
// ---------------------------------------------------------------------------
__global__ __launch_bounds__(256)
void scan_kernel(const float* __restrict__ xin,   // (B*L, D) silu'd input
                 const float* __restrict__ dt,    // (B*L, D)
                 const float* __restrict__ xp,    // (B*L, 112): [48+2n]=B, [80+2n]=C
                 const float* __restrict__ zs,    // (B*L, D) silu(z)
                 const float* __restrict__ A_log, // (16)
                 __bf16*      __restrict__ ybf)   // (B*L, D) y*silu(z), bf16
{
    const int tid = blockIdx.x * 256 + threadIdx.x;  // 49152 threads
    const int n   = tid & 15;
    const int bd  = tid >> 4;                        // 0..3071
    const int d   = bd % D_;
    const int b   = bd / D_;

    const float Aneg  = -__expf(A_log[n]);
    const float phase = 6.28318530717958647692f * (float)n * (1.0f / 16.0f);
    float h0 = 0.01f * __cosf(phase);
    float h1 = 0.01f * __sinf(phase);

    for (int t = 0; t < L_; ++t) {
        const size_t base = (size_t)(b * L_ + t);
        const float dtt = dt [base * D_ + d];
        const float xt  = xin[base * D_ + d];
        const float2 Bt = *(const float2*)(xp + base * XPW + R_      + 2 * n);
        const float2 Ct = *(const float2*)(xp + base * XPW + R_ + 32 + 2 * n);

        // fractal compression: h <- |h|^e * h/|h|
        const float alpha = __expf(dtt * Aneg);
        const float e     = 1.0f + 0.5f * (1.0f - alpha);
        const float mag   = __builtin_sqrtf(h0 * h0 + h1 * h1 + 1e-8f);
        const float s1    = __powf(mag, e) / mag;
        h0 *= s1; h1 *= s1;

        // input injection
        const float xd = xt * dtt;
        h0 = fmaf(Bt.x, xd, h0);
        h1 = fmaf(Bt.y, xd, h1);

        // tanh magnitude squash: h <- tanh(|h|) * h/|h|
        const float m2 = __builtin_sqrtf(h0 * h0 + h1 * h1 + 1e-8f);
        const float th = 1.0f - 2.0f / (__expf(2.0f * m2) + 1.0f);
        const float s2 = th / m2;
        h0 *= s2; h1 *= s2;

        // y = sum_n Ct . h   (reduce 16 lanes of the half-wave)
        float y = Ct.x * h0 + Ct.y * h1;
        y += __shfl_xor(y, 1, 16);
        y += __shfl_xor(y, 2, 16);
        y += __shfl_xor(y, 4, 16);
        y += __shfl_xor(y, 8, 16);

        if (n == 0) {
            const float z = zs[base * D_ + d];
            ybf[base * D_ + d] = (__bf16)(y * z);
        }
    }
}

// ---------------------------------------------------------------------------
extern "C" void kernel_launch(void* const* d_in, const int* in_sizes, int n_in,
                              void* d_out, int out_size, void* d_ws, size_t ws_size,
                              hipStream_t stream)
{
    const float* x    = (const float*)d_in[0];   // (4,1024,768)
    const float* wIn  = (const float*)d_in[1];   // (768,1536)
    const float* wXp  = (const float*)d_in[2];   // (768,112)
    const float* wDt  = (const float*)d_in[3];   // (48,768)
    const float* bDt  = (const float*)d_in[4];   // (768)
    const float* wOut = (const float*)d_in[5];   // (768,768)
    const float* bOut = (const float*)d_in[6];   // (768)
    const float* Alog = (const float*)d_in[7];   // (16)
    float* out = (float*)d_out;                  // (4,1024,768) f32

    // ---- workspace carve (256B aligned) ----
    char* p = (char*)d_ws;
    auto carve = [&](size_t bytes) -> void* {
        void* r = (void*)p;
        p += (bytes + 255) & ~(size_t)255;
        return r;
    };
    __bf16* xb   = (__bf16*)carve((size_t)MTOT * D_   * 2);
    __bf16* wib  = (__bf16*)carve((size_t)D_   * 1536 * 2);
    __bf16* wxb  = (__bf16*)carve((size_t)D_   * XPWP * 2);  // padded cols
    __bf16* wdb  = (__bf16*)carve((size_t)KDTP * D_   * 2);  // padded rows
    __bf16* wob  = (__bf16*)carve((size_t)D_   * D_   * 2);
    float*  xinf = (float*) carve((size_t)MTOT * D_   * 4);
    __bf16* xinb = (__bf16*)carve((size_t)MTOT * D_   * 2);
    float*  zsf  = (float*) carve((size_t)MTOT * D_   * 4);
    float*  xpf  = (float*) carve((size_t)MTOT * XPW  * 4);
    __bf16* xpb  = (__bf16*)carve((size_t)MTOT * XPWP * 2);  // padded lda
    float*  dtf  = (float*) carve((size_t)MTOT * D_   * 4);
    __bf16* yb   = (__bf16*)carve((size_t)MTOT * D_   * 2);

    auto cvt = [&](const float* s, __bf16* dst, int sr, int sc, int dr, int dc) {
        int total = dr * dc;
        pad_convert_kernel<<<(total + 255) / 256, 256, 0, stream>>>(
            s, dst, sr, sc, dc, total);
    };
    cvt(x,    xb,  MTOT, D_,   MTOT, D_);
    cvt(wIn,  wib, D_,   1536, D_,   1536);
    cvt(wXp,  wxb, D_,   XPW,  D_,   XPWP);   // zero-pad cols 112..127
    cvt(wDt,  wdb, R_,   D_,   KDTP, D_);     // zero-pad rows 48..63
    cvt(wOut, wob, D_,   D_,   D_,   D_);

    // K1: xz = x @ in_proj_w  -> silu -> x_in (f32+bf16), zsilu
    gemm_bf16_wmma<0><<<dim3(1536 / 64, MTOT / 128), 256, 0, stream>>>(
        xb, D_, wib, 1536, D_, nullptr, xinf, xinb, zsf);

    // K2: xp = x_in @ x_proj_w (N padded to 128)
    gemm_bf16_wmma<1><<<dim3(XPWP / 64, MTOT / 128), 256, 0, stream>>>(
        xinb, D_, wxb, XPWP, D_, nullptr, xpf, xpb, nullptr);

    // K3: dt = softplus(xp[:, :48] @ dt_proj_w + b)  (K padded to 64)
    gemm_bf16_wmma<2><<<dim3(D_ / 64, MTOT / 128), 256, 0, stream>>>(
        xpb, XPWP, wdb, D_, KDTP, bDt, dtf, nullptr, nullptr);

    // K4: sequential scan, 49152 threads (1536 wave32 waves)
    scan_kernel<<<(B_ * D_ * N_) / 256, 256, 0, stream>>>(
        xinf, dtf, xpf, zsf, Alog, yb);

    // K5: out = y @ out_proj_w + b
    gemm_bf16_wmma<3><<<dim3(D_ / 64, MTOT / 128), 256, 0, stream>>>(
        yb, D_, wob, D_, D_, bOut, out, nullptr, nullptr);
}
